// Conv2d_45810121179422
// MI455X (gfx1250) — matrix operand order repaired and checked by reading, not compiled, not run on hardware
//
#include <hip/hip_runtime.h>

typedef __attribute__((ext_vector_type(16))) __bf16 v16bf;
typedef __attribute__((ext_vector_type(8)))  float  v8f;

#define IN_C  128
#define OUT_C 256
#define HW    56
#define HW2   3136      // 56*56
#define HP    58        // padded H/W
#define MT    128       // oc per block
#define NT    128       // pixels per block
#define KT    32        // K per step (k = (kh*3+kw)*128 + ic), 36 steps

// workspace layout
#define WPACK_ELEMS  (2u*36u*2u*128u*32u)          // 589824 bf16 = 1,179,648 B
#define WPACK_BYTES  (WPACK_ELEMS*2u)
#define XPLANE_ELEMS ((size_t)32*HP*HP*IN_C)       // 13,778,944 bf16
#define XPLANE_BYTES (XPLANE_ELEMS*2)

// ---------------------------------------------------------------------------
// async global->LDS copy (CDNA5, ASYNCcnt-tracked; bypasses VGPRs)
__device__ __forceinline__ void async_b128(unsigned lds_byte_off, const void* gsrc) {
    asm volatile("global_load_async_to_lds_b128 %0, %1, off"
                 :: "v"(lds_byte_off), "v"((unsigned long long)(uintptr_t)gsrc)
                 : "memory");
}

// ---------------------------------------------------------------------------
// Prologue 1: pack weights -> bf16 hi/lo in the exact LDS tile image:
// wpack[ocb][ks][plane][ocl][kk]   (16KB contiguous per (ocb,ks))
__global__ __launch_bounds__(256) void pack_w_kernel(const float* __restrict__ wgt,
                                                     __bf16* __restrict__ wpack) {
    const int idx  = blockIdx.x * 256 + threadIdx.x;    // 0 .. 294911
    const int kk   = idx & 31;
    const int ocl  = (idx >> 5) & 127;
    const int rest = idx >> 12;                         // 0..71
    const int ks   = rest % 36;
    const int ocb  = rest / 36;
    const int pos  = ks >> 2;                           // kh*3+kw
    const int ic   = ((ks & 3) << 5) + kk;
    const int oc   = ocb * 128 + ocl;
    const float v  = wgt[(size_t)(oc * IN_C + ic) * 9 + pos];
    const __bf16 hi = (__bf16)v;
    const __bf16 lo = (__bf16)(v - (float)hi);
    const size_t base = ((size_t)(ocb * 36 + ks) * 2) * 4096 + ocl * 32 + kk;
    wpack[base]        = hi;     // plane 0
    wpack[base + 4096] = lo;     // plane 1
}

// ---------------------------------------------------------------------------
// Prologue 2: NCHW fp32 -> padded NHWC bf16 hi/lo planes (LDS tile transpose).
// xs[n][hp][wp][ic], borders pre-zeroed by memset. One block per (n,h) row.
__global__ __launch_bounds__(256) void split_x_kernel(const float* __restrict__ x,
                                                      __bf16* __restrict__ xhi,
                                                      __bf16* __restrict__ xlo) {
    __shared__ float tile[IN_C][HW];                    // 28,672 B
    const int b = blockIdx.x;
    const int n = b / HW;
    const int h = b - n * HW;
    const int t = threadIdx.x;

    // coalesced read: lanes sweep w, 4 ic rows per pass
    const int wr  = t & 63;
    const int icr = t >> 6;
    if (wr < HW) {
        #pragma unroll
        for (int i = 0; i < 32; ++i) {
            const int ic = icr + i * 4;
            tile[ic][wr] = x[(((size_t)n * IN_C + ic) * HW + h) * HW + wr];
        }
    }
    __syncthreads();

    // coalesced write: lanes sweep ic (contiguous in NHWC)
    const int icw = t & 127;
    const int w2  = t >> 7;                             // 2 w per pass
    const size_t rowbase = (((size_t)n * HP + (h + 1)) * HP + 1) * IN_C;
    #pragma unroll
    for (int i = 0; i < 28; ++i) {
        const int w = w2 + i * 2;
        const float v = tile[icw][w];
        const __bf16 hi = (__bf16)v;
        const __bf16 lo = (__bf16)(v - (float)hi);
        xhi[rowbase + (size_t)w * IN_C + icw] = hi;
        xlo[rowbase + (size_t)w * IN_C + icw] = lo;
    }
}

// ---------------------------------------------------------------------------
// Main kernel: implicit GEMM, bf16x3 WMMA, async-DMA double-buffered staging.
__global__ __launch_bounds__(256) void conv3x3_wmma_async(
    const __bf16* __restrict__ wpack, const __bf16* __restrict__ xhi,
    const __bf16* __restrict__ xlo, const float* __restrict__ bias,
    float* __restrict__ out)
{
    // [buf][plane][row][kk] ; buf stride 8192 elems, plane stride 4096 elems
    __shared__ __bf16 sA[2 * 2 * 128 * 32];   // 32 KB
    __shared__ __bf16 sB[2 * 2 * 128 * 32];   // 32 KB

    const int t    = threadIdx.x;
    const int lane = t & 31;
    const int wave = t >> 5;
    const int wm   = wave & 1;
    const int wn   = wave >> 1;
    const int lr   = lane & 15;
    const int lhi  = lane >> 4;

    const int p0  = blockIdx.x * NT;
    const int ocb = blockIdx.y;
    const int oc0 = ocb * MT;

    // per-thread B-copy decode (constant over K loop)
    const int pl   = t >> 1;
    const int half = t & 1;
    const int p    = p0 + pl;
    const int pn   = p / HW2;
    const int prem = p - pn * HW2;
    const int oh   = prem / HW;
    const int ow   = prem - oh * HW;

    const unsigned ldsA = (unsigned)(uintptr_t)(void*)sA;
    const unsigned ldsB = (unsigned)(uintptr_t)(void*)sB;

    // issue one K-step's staging (8 async b128 per thread)
    auto issue = [&](int ks, int buf) {
        const int pos = ks >> 2;
        const int kh  = pos / 3;
        const int kw  = pos - kh * 3;
        const int ic0 = (ks & 3) << 5;
        // A: 16KB contiguous (both planes pre-packed)
        {
            const char* src = (const char*)(wpack + ((size_t)(ocb * 36 + ks) << 13))
                              + t * 64;
            const unsigned dst = ldsA + (unsigned)buf * 16384u + (unsigned)t * 64u;
            async_b128(dst +  0, src +  0);
            async_b128(dst + 16, src + 16);
            async_b128(dst + 32, src + 32);
            async_b128(dst + 48, src + 48);
        }
        // B: per-pixel 64B contiguous K-slices from padded NHWC planes (no bounds checks)
        {
            const size_t eoff = (((size_t)pn * HP + (oh + kh)) * HP + (ow + kw)) * IN_C + ic0;
            const char* shi = (const char*)(xhi + eoff) + half * 32;
            const char* slo = (const char*)(xlo + eoff) + half * 32;
            const unsigned d0 = ldsB + (unsigned)buf * 16384u + (unsigned)pl * 64u
                                + (unsigned)half * 32u;
            async_b128(d0,          shi);
            async_b128(d0 + 16,     shi + 16);
            async_b128(d0 + 8192,      slo);       // lo plane
            async_b128(d0 + 8192 + 16, slo + 16);
        }
    };

    v8f acc[4][2];
    #pragma unroll
    for (int i = 0; i < 4; ++i)
        #pragma unroll
        for (int j = 0; j < 2; ++j)
            #pragma unroll
            for (int r = 0; r < 8; ++r) acc[i][j][r] = 0.0f;

    issue(0, 0);

    for (int ks = 0; ks < 36; ++ks) {
        const int buf = ks & 1;
        if (ks < 35) {
            issue(ks + 1, buf ^ 1);
            asm volatile("s_wait_asynccnt 0x8" ::: "memory");  // batch ks landed
        } else {
            asm volatile("s_wait_asynccnt 0x0" ::: "memory");
        }
        __syncthreads();   // all waves' DMA for this buf visible

        // fragments per ISA wave32 layouts (two ds_load_b128 each)
        const __bf16* Ah = sA + buf * 8192;          // plane hi
        const __bf16* Al = sA + buf * 8192 + 4096;   // plane lo
        const __bf16* Bh = sB + buf * 8192;
        const __bf16* Bl = sB + buf * 8192 + 4096;

        v16bf ah[4], al[4], bh[2], bl[2];
        #pragma unroll
        for (int mt = 0; mt < 4; ++mt) {
            const int row  = wm * 64 + mt * 16 + lr;
            const int base = row * KT + lhi * 8;
            ((uint4*)&ah[mt])[0] = *(const uint4*)(Ah + base);
            ((uint4*)&ah[mt])[1] = *(const uint4*)(Ah + base + 16);
            ((uint4*)&al[mt])[0] = *(const uint4*)(Al + base);
            ((uint4*)&al[mt])[1] = *(const uint4*)(Al + base + 16);
        }
        #pragma unroll
        for (int nt = 0; nt < 2; ++nt) {
            const int col  = wn * 32 + nt * 16 + lr;
            const int base = col * KT + lhi * 8;
            ((uint4*)&bh[nt])[0] = *(const uint4*)(Bh + base);
            ((uint4*)&bh[nt])[1] = *(const uint4*)(Bh + base + 16);
            ((uint4*)&bl[nt])[0] = *(const uint4*)(Bl + base);
            ((uint4*)&bl[nt])[1] = *(const uint4*)(Bl + base + 16);
        }

        // bf16x3: hi*hi + hi*lo + lo*hi, fp32 accumulate
        #pragma unroll
        for (int mt = 0; mt < 4; ++mt)
            #pragma unroll
            for (int nt = 0; nt < 2; ++nt) {
                acc[mt][nt] = __builtin_amdgcn_wmma_f32_16x16x32_bf16(
                    false, ah[mt], false, bh[nt], (short)0, acc[mt][nt], false, false);
                acc[mt][nt] = __builtin_amdgcn_wmma_f32_16x16x32_bf16(
                    false, ah[mt], false, bl[nt], (short)0, acc[mt][nt], false, false);
                acc[mt][nt] = __builtin_amdgcn_wmma_f32_16x16x32_bf16(
                    false, al[mt], false, bh[nt], (short)0, acc[mt][nt], false, false);
            }
        __syncthreads();   // reads of this buf done before it is re-filled
    }

    // epilogue: C frag (VGPR r -> M = r + 8*lhi, N = lr), fused bias, NCHW store
    #pragma unroll
    for (int nt = 0; nt < 2; ++nt) {
        const int pout = p0 + wn * 32 + nt * 16 + lr;
        const int n    = pout / HW2;
        const int rem  = pout - n * HW2;
        float* ob = out + (size_t)n * OUT_C * HW2 + rem;
        #pragma unroll
        for (int mt = 0; mt < 4; ++mt) {
            const int ocbase = oc0 + wm * 64 + mt * 16 + lhi * 8;
            #pragma unroll
            for (int r = 0; r < 8; ++r) {
                const int oc = ocbase + r;
                ob[(size_t)oc * HW2] = acc[mt][nt][r] + bias[oc];
            }
        }
    }
}

// ---------------------------------------------------------------------------
// Fallback (round-1 kernel): in-kernel fp32 staging + split, no workspace.
__global__ __launch_bounds__(256) void conv3x3_wmma_bf16x3(
    const float* __restrict__ x, const float* __restrict__ wgt,
    const float* __restrict__ bias, float* __restrict__ out)
{
    __shared__ __bf16 sA_hi[MT * KT];
    __shared__ __bf16 sA_lo[MT * KT];
    __shared__ __bf16 sB_hi[NT * KT];
    __shared__ __bf16 sB_lo[NT * KT];

    const int t    = threadIdx.x;
    const int lane = t & 31;
    const int wave = t >> 5;
    const int wm   = wave & 1;
    const int wn   = wave >> 1;
    const int lr   = lane & 15;
    const int lhi  = lane >> 4;

    const int p0  = blockIdx.x * NT;
    const int oc0 = blockIdx.y * MT;

    const int pl   = t & 127;
    const int kk0  = t >> 7;
    const int p    = p0 + pl;
    const int pn   = p / HW2;
    const int prem = p - pn * HW2;
    const int oh   = prem / HW;
    const int ow   = prem - oh * HW;

    const int akk  = t & 31;
    const int aoc0 = t >> 5;

    v8f acc[4][2];
    #pragma unroll
    for (int i = 0; i < 4; ++i)
        #pragma unroll
        for (int j = 0; j < 2; ++j)
            #pragma unroll
            for (int r = 0; r < 8; ++r) acc[i][j][r] = 0.0f;

    for (int ks = 0; ks < 36; ++ks) {
        const int k0  = ks * KT;
        const int pos = k0 >> 7;
        const int ic0 = k0 & 127;
        const int kh  = pos / 3;
        const int kw  = pos - kh * 3;
        {
            const int ic = ic0 + akk;
            #pragma unroll
            for (int i = 0; i < 16; ++i) {
                const int ocl = aoc0 + i * 8;
                const float v = wgt[(size_t)((oc0 + ocl) * IN_C + ic) * 9 + pos];
                const __bf16 hi = (__bf16)v;
                const __bf16 lo = (__bf16)(v - (float)hi);
                sA_hi[ocl * KT + akk] = hi;
                sA_lo[ocl * KT + akk] = lo;
            }
        }
        {
            const int ih = oh - 1 + kh;
            const int iw = ow - 1 + kw;
            const bool inb = (ih >= 0) & (ih < HW) & (iw >= 0) & (iw < HW);
            const float* xb = x + (size_t)pn * IN_C * HW2 + ih * HW + iw;
            #pragma unroll
            for (int i = 0; i < 16; ++i) {
                const int kk = kk0 + i * 2;
                const int ic = ic0 + kk;
                const float v = inb ? xb[(size_t)ic * HW2] : 0.0f;
                const __bf16 hi = (__bf16)v;
                const __bf16 lo = (__bf16)(v - (float)hi);
                sB_hi[pl * KT + kk] = hi;
                sB_lo[pl * KT + kk] = lo;
            }
        }
        __syncthreads();

        v16bf ah[4], al[4], bh[2], bl[2];
        #pragma unroll
        for (int mt = 0; mt < 4; ++mt) {
            const int row  = wm * 64 + mt * 16 + lr;
            const int base = row * KT + lhi * 8;
            ((uint4*)&ah[mt])[0] = *(const uint4*)(sA_hi + base);
            ((uint4*)&ah[mt])[1] = *(const uint4*)(sA_hi + base + 16);
            ((uint4*)&al[mt])[0] = *(const uint4*)(sA_lo + base);
            ((uint4*)&al[mt])[1] = *(const uint4*)(sA_lo + base + 16);
        }
        #pragma unroll
        for (int nt = 0; nt < 2; ++nt) {
            const int col  = wn * 32 + nt * 16 + lr;
            const int base = col * KT + lhi * 8;
            ((uint4*)&bh[nt])[0] = *(const uint4*)(sB_hi + base);
            ((uint4*)&bh[nt])[1] = *(const uint4*)(sB_hi + base + 16);
            ((uint4*)&bl[nt])[0] = *(const uint4*)(sB_lo + base);
            ((uint4*)&bl[nt])[1] = *(const uint4*)(sB_lo + base + 16);
        }

        #pragma unroll
        for (int mt = 0; mt < 4; ++mt)
            #pragma unroll
            for (int nt = 0; nt < 2; ++nt) {
                acc[mt][nt] = __builtin_amdgcn_wmma_f32_16x16x32_bf16(
                    false, ah[mt], false, bh[nt], (short)0, acc[mt][nt], false, false);
                acc[mt][nt] = __builtin_amdgcn_wmma_f32_16x16x32_bf16(
                    false, ah[mt], false, bl[nt], (short)0, acc[mt][nt], false, false);
                acc[mt][nt] = __builtin_amdgcn_wmma_f32_16x16x32_bf16(
                    false, al[mt], false, bh[nt], (short)0, acc[mt][nt], false, false);
            }
        __syncthreads();
    }

    #pragma unroll
    for (int nt = 0; nt < 2; ++nt) {
        const int pout = p0 + wn * 32 + nt * 16 + lr;
        const int n    = pout / HW2;
        const int rem  = pout - n * HW2;
        float* ob = out + (size_t)n * OUT_C * HW2 + rem;
        #pragma unroll
        for (int mt = 0; mt < 4; ++mt) {
            const int ocbase = oc0 + wm * 64 + mt * 16 + lhi * 8;
            #pragma unroll
            for (int r = 0; r < 8; ++r) {
                const int oc = ocbase + r;
                ob[(size_t)oc * HW2] = acc[mt][nt][r] + bias[oc];
            }
        }
    }
}

// ---------------------------------------------------------------------------
extern "C" void kernel_launch(void* const* d_in, const int* in_sizes, int n_in,
                              void* d_out, int out_size, void* d_ws, size_t ws_size,
                              hipStream_t stream) {
    (void)in_sizes; (void)n_in; (void)out_size;
    const float* x    = (const float*)d_in[0];
    const float* wgt  = (const float*)d_in[1];
    const float* bias = (const float*)d_in[2];
    float* out        = (float*)d_out;

    const size_t ws_need = (size_t)WPACK_BYTES + 2 * XPLANE_BYTES;  // ~56.3 MB
    if (ws_size >= ws_need) {
        __bf16* wpack = (__bf16*)d_ws;
        __bf16* xhi   = (__bf16*)((char*)d_ws + WPACK_BYTES);
        __bf16* xlo   = (__bf16*)((char*)d_ws + WPACK_BYTES + XPLANE_BYTES);
        // zero padded planes (borders), stream op => graph-capturable
        hipMemsetAsync(xhi, 0, 2 * XPLANE_BYTES, stream);
        pack_w_kernel <<<294912 / 256, 256, 0, stream>>>(wgt, wpack);
        split_x_kernel<<<32 * HW,      256, 0, stream>>>(x, xhi, xlo);
        conv3x3_wmma_async<<<dim3(784, 2), 256, 0, stream>>>(wpack, xhi, xlo, bias, out);
    } else {
        conv3x3_wmma_bf16x3<<<dim3(784, 2), 256, 0, stream>>>(x, wgt, bias, out);
    }
}
